// OneScaleAttentionLayer_79156247265568
// MI455X (gfx1250) — matrix operand order repaired and checked by reading, not compiled, not run on hardware
//
#include <hip/hip_runtime.h>

#define NB 32
#define LQ 1024
#define LA 1024
#define HD 512
#define NEGV -1e7f
#define TILE_U4 (32 * HD * 2 / 16)   // 2048 16-byte chunks per 32xHD bf16 tile

typedef __attribute__((ext_vector_type(16))) __bf16 v16bf;
typedef __attribute__((ext_vector_type(8)))  float  v8f;
typedef unsigned int u32x4 __attribute__((ext_vector_type(4)));
typedef int          i32x8 __attribute__((ext_vector_type(8)));
typedef int          i32x4 __attribute__((ext_vector_type(4)));
typedef int          v4i   __attribute__((ext_vector_type(4)));
typedef __attribute__((address_space(1))) v4i gv4i;   // global int4
typedef __attribute__((address_space(3))) v4i lv4i;   // LDS int4

#ifndef TRY_TDM
#define TRY_TDM 1
#endif

#if defined(__has_builtin)
#  if __has_builtin(__builtin_amdgcn_global_load_async_to_lds_b128) && \
      __has_builtin(__builtin_amdgcn_s_wait_asynccnt)
#    define HAS_ASYNC 1
#  else
#    define HAS_ASYNC 0
#  endif
#  if TRY_TDM && __has_builtin(__builtin_amdgcn_tensor_load_to_lds) && \
      __has_builtin(__builtin_amdgcn_s_wait_tensorcnt)
#    define HAS_TDM 1
#  else
#    define HAS_TDM 0
#  endif
#else
#  define HAS_ASYNC 0
#  define HAS_TDM 0
#endif

__device__ inline __bf16 f2bf(float f) {
  unsigned u = __float_as_uint(f);
  unsigned r = (u + 0x7fffu + ((u >> 16) & 1u)) >> 16;   // RNE
  unsigned short s = (unsigned short)r;
  __bf16 b;
  __builtin_memcpy(&b, &s, sizeof(b));
  return b;
}

// ---------------------------------------------------------------------------
// Tile staging: 32 rows x HD bf16 (32 KB) from global into LDS.
//   stage_tdm  : one Tensor-Data-Mover descriptor (ISA 08_async_tensor §8.3/8.4)
//   stage_async: GLOBAL_LOAD_ASYNC_TO_LDS_B128, tracked by ASYNCcnt
//   stage_plain: VALU copy fallback
// Caller must __syncthreads() after staging.
// ---------------------------------------------------------------------------
__device__ inline void stage_plain(__bf16* lds, const __bf16* gsrc, int tid) {
  const uint4* s = (const uint4*)gsrc;
  uint4* d = (uint4*)lds;
  for (int i = tid; i < TILE_U4; i += 128) d[i] = s[i];
}

#if HAS_TDM
__device__ inline void stage_tdm(__bf16* lds, const __bf16* gsrc, int rows_total, int tid) {
  if (tid < 32) {   // one wave issues the DMA (EXEC-independent, per-wave op)
    unsigned lds_off = (unsigned)(unsigned long long)(__attribute__((address_space(3))) void*)lds;
    unsigned long long ga = (unsigned long long)(const void*)gsrc;
    // D# group0: count=1 | lds_addr | global_addr[56:0] | type=2
    u32x4 g0 = { 1u, lds_off, (unsigned)ga,
                 (unsigned)((ga >> 32) & 0x01ffffffu) | (2u << 30) };
    // D# group1: data_size=2B; tensor_dim0=HD; tensor_dim1=rows_total;
    //            tile_dim0=HD; tile_dim1=32; tensor_dim0_stride=HD
    i32x8 g1 = { (int)(1u << 16),
                 (int)((unsigned)HD << 16),
                 (int)((unsigned)(rows_total & 0xffff) << 16),
                 (int)((unsigned)HD << 16),
                 32,
                 HD,
                 0, 0 };
    i32x4 g2 = { 0, 0, 0, 0 };
    i32x4 g3 = { 0, 0, 0, 0 };
    i32x8 g4 = { 0, 0, 0, 0, 0, 0, 0, 0 };   // 6-arg toolchain form: extra group, zero-filled
    __builtin_amdgcn_tensor_load_to_lds(g0, g1, g2, g3, g4, 0);
    __builtin_amdgcn_s_wait_tensorcnt(0);
  }
}
#endif

#if HAS_ASYNC
__device__ inline void stage_async(__bf16* lds, const __bf16* gsrc, int tid) {
  for (int i = tid; i < TILE_U4; i += 128) {
    __builtin_amdgcn_global_load_async_to_lds_b128(
        (gv4i*)((const uint4*)gsrc + i),
        (lv4i*)((uint4*)lds + i),
        0, 0);
  }
  __builtin_amdgcn_s_wait_asynccnt(0);
}
#endif

__device__ inline void stage_A(__bf16* lds, const __bf16* gsrc, int rows_total, int tid) {
#if HAS_TDM
  stage_tdm(lds, gsrc, rows_total, tid);
#elif HAS_ASYNC
  (void)rows_total; stage_async(lds, gsrc, tid);
#else
  (void)rows_total; stage_plain(lds, gsrc, tid);
#endif
}

__device__ inline void stage_B(__bf16* lds, const __bf16* gsrc, int rows_total, int tid) {
#if HAS_ASYNC
  (void)rows_total; stage_async(lds, gsrc, tid);
#elif HAS_TDM
  stage_tdm(lds, gsrc, rows_total, tid);
#else
  (void)rows_total; stage_plain(lds, gsrc, tid);
#endif
}

// ---------------------------------------------------------------------------
// Kernel 1: convert a -> bf16, build U^T in bf16 (contiguous-K B fragments).
// ---------------------------------------------------------------------------
__global__ void prep_kernel(const float* __restrict__ a, const float* __restrict__ U,
                            __bf16* __restrict__ a_bf, __bf16* __restrict__ Ut_bf) {
  size_t stride = (size_t)gridDim.x * blockDim.x;
  size_t t0 = (size_t)blockIdx.x * blockDim.x + threadIdx.x;
  const size_t Na = (size_t)NB * LA * HD;
  for (size_t i = t0; i < Na; i += stride) a_bf[i] = f2bf(a[i]);
  const size_t Nu = (size_t)HD * HD;
  for (size_t i = t0; i < Nu; i += stride) {
    size_t n = i / HD, k = i % HD;
    Ut_bf[n * HD + k] = f2bf(U[k * HD + n]);
  }
}

// ---------------------------------------------------------------------------
// WMMA fragment loaders (V_WMMA_F32_16X16X32_BF16, wave32; A/B in this code's contiguous-K order, C hw):
//   A (16x32, MxK): lane m = lane&15; e   -> K=k0+16*hi+e
//   B (32x16, KxN): lane n = lane&15; e   -> K=k0+16*hi+e   (row-striped)
//   C (16x16)     : lane n = lane&15; VGPR r -> row r+8*hi
// ---------------------------------------------------------------------------
__device__ inline v16bf load_a_frag(const __bf16* __restrict__ Arow, int k0, int hi) {
  v16bf af;
  const __bf16* ap = Arow + k0 + 16 * hi;
#pragma unroll
  for (int e = 0; e < 8; ++e)  af[e] = ap[e];
#pragma unroll
  for (int e = 8; e < 16; ++e) af[e] = ap[e];
  return af;
}

__device__ inline v16bf load_b_frag(const __bf16* __restrict__ Btrow, int k0, int hi) {
  v16bf bf;
  const __bf16* bp = Btrow + k0 + 16 * hi;
#pragma unroll
  for (int e = 0; e < 16; ++e) bf[e] = bp[e];
  return bf;
}

// ---------------------------------------------------------------------------
// Kernel 2: qU[b] = q[b] @ U (bf16 out). 128 thr = 4 waves = 2(M)x2(N).
// A tile converted to LDS once; B tile DMA-staged per N-step; C tile bounced
// through LDS for coalesced b128 stores.
// ---------------------------------------------------------------------------
__global__ __launch_bounds__(128)
void gemm_qU_kernel(const float* __restrict__ q, const __bf16* __restrict__ Ut,
                    __bf16* __restrict__ qU) {
  __shared__ __align__(16) __bf16 As[32 * HD];
  __shared__ __align__(16) __bf16 Bs[32 * HD];
  __bf16* Cs = Bs;   // C tile reuses Bs after the K-loop of each N-step
  int b  = blockIdx.x >> 5;
  int m0 = (blockIdx.x & 31) * 32;
  int tid = threadIdx.x;
  const float* qrow = q + ((size_t)b * LQ + m0) * HD;
  for (int idx = tid; idx < 32 * HD; idx += 128) As[idx] = f2bf(qrow[idx]);

  int lane = tid & 31, wave = tid >> 5;
  int waveM = wave >> 1, waveN = wave & 1;
  int hi = lane >> 4, col = lane & 15;
  const __bf16* Arow = &As[(waveM * 16 + col) * HD];

  for (int n0 = 0; n0 < HD; n0 += 32) {
    __syncthreads();                                    // Bs/Cs safe to overwrite
    stage_B(Bs, Ut + (size_t)n0 * HD, HD, tid);         // rows n0..n0+31 of U^T
    __syncthreads();
    const __bf16* Btrow = &Bs[(waveN * 16 + col) * HD];
    v8f c = {};
    for (int k0 = 0; k0 < HD; k0 += 32) {
      v16bf af = load_a_frag(Arow, k0, hi);
      v16bf bf = load_b_frag(Btrow, k0, hi);
      c = __builtin_amdgcn_wmma_f32_16x16x32_bf16(false, af, false, bf,
                                                  (short)0, c, false, false);
    }
    __syncthreads();                                    // Bs reads done -> Cs ok
#pragma unroll
    for (int r = 0; r < 8; ++r)
      Cs[(waveM * 16 + r + 8 * hi) * 32 + waveN * 16 + col] = f2bf(c[r]);
    __syncthreads();
    int rr = tid >> 2, cc = tid & 3;                    // 32 rows x 64B, 1 b128/thr
    ((uint4*)(qU + ((size_t)b * LQ + m0 + rr) * HD + n0))[cc] =
        ((const uint4*)&Cs[(size_t)rr * 32])[cc];
  }
}

// ---------------------------------------------------------------------------
// Kernel 3: X tile = qU_tile @ a^T, fused mask->sigmoid->exp with row/col
// (sum,max) statistics. Deterministic: exclusive partial buffers, no atomics.
// ---------------------------------------------------------------------------
__global__ __launch_bounds__(128)
void score_kernel(const __bf16* __restrict__ qU, const __bf16* __restrict__ a_bf,
                  const float* __restrict__ q_mask, const float* __restrict__ a_mask,
                  float* __restrict__ rowpsum, float* __restrict__ rowpmax,
                  float* __restrict__ colpsum, float* __restrict__ colpmax) {
  __shared__ __align__(16) __bf16 As[32 * HD];
  __shared__ __align__(16) __bf16 Bs[32 * HD];
  int b  = blockIdx.x >> 5;
  int mt = blockIdx.x & 31;
  int m0 = mt * 32;
  int tid = threadIdx.x;
  stage_A(As, qU + ((size_t)b * LQ + m0) * HD, LQ, tid);   // TDM if available

  int lane = tid & 31, wave = tid >> 5;
  int waveM = wave >> 1, waveN = wave & 1;
  int hi = lane >> 4, col = lane & 15;
  const __bf16* Arow = &As[(waveM * 16 + col) * HD];

  float qm[8];
#pragma unroll
  for (int r = 0; r < 8; ++r)
    qm[r] = q_mask[(size_t)b * LQ + m0 + waveM * 16 + r + 8 * hi];

  float rowS[8], rowM[8];
#pragma unroll
  for (int r = 0; r < 8; ++r) { rowS[r] = 0.f; rowM[r] = 0.f; }

  for (int n0 = 0; n0 < LA; n0 += 32) {
    __syncthreads();                                       // Bs safe to overwrite
    stage_B(Bs, a_bf + ((size_t)b * LA + n0) * HD, LA, tid);
    __syncthreads();
    int ng = n0 + waveN * 16 + col;
    const __bf16* Btrow = &Bs[(waveN * 16 + col) * HD];
    float am = a_mask[(size_t)b * LA + ng];
    v8f c = {};
    for (int k0 = 0; k0 < HD; k0 += 32) {
      v16bf af = load_a_frag(Arow, k0, hi);
      v16bf bf = load_b_frag(Btrow, k0, hi);
      c = __builtin_amdgcn_wmma_f32_16x16x32_bf16(false, af, false, bf,
                                                  (short)0, c, false, false);
    }
    float colS = 0.f, colM = 0.f;
#pragma unroll
    for (int r = 0; r < 8; ++r) {
      float x = c[r];
      if (!(qm[r] * am > 0.f)) x = NEGV;
      float I  = 1.f / (1.f + __expf(-x));   // sigmoid; I in (0,1)
      float ev = __expf(I);                  // bounded in (1,e): no max-shift needed
      colS += ev;
      colM = fmaxf(colM, I);
      float rs = ev, rm = I;
#pragma unroll
      for (int off = 1; off <= 8; off <<= 1) {   // reduce across 16 cols (half-wave)
        rs += __shfl_xor(rs, off);
        rm = fmaxf(rm, __shfl_xor(rm, off));
      }
      rowS[r] += rs;
      rowM[r] = fmaxf(rowM[r], rm);
    }
    colS += __shfl_xor(colS, 16);             // combine 8-row halves -> 16 rows
    colM = fmaxf(colM, __shfl_xor(colM, 16));
    if (lane < 16) {
      int wr = mt * 2 + waveM;                // 16-row stripe id (0..63)
      size_t ci = ((size_t)b * 64 + wr) * LA + ng;
      colpsum[ci] = colS;
      colpmax[ci] = colM;
    }
  }

  if (col == 0) {   // one writer per (row, half); waveN indexes the partial slot
#pragma unroll
    for (int r = 0; r < 8; ++r) {
      int row = m0 + waveM * 16 + r + 8 * hi;
      size_t ri = ((size_t)b * LQ + row) * 2 + waveN;
      rowpsum[ri] = rowS[r];
      rowpmax[ri] = rowM[r];
    }
  }
}

// ---------------------------------------------------------------------------
// Kernel 4: att = exp(max)/sumexp; q_attn = att_a2q @ q, a_attn = att_q2a @ a
// ---------------------------------------------------------------------------
__global__ __launch_bounds__(256)
void finish_kernel(const float* __restrict__ q, const float* __restrict__ a,
                   const float* __restrict__ rowpsum, const float* __restrict__ rowpmax,
                   const float* __restrict__ colpsum, const float* __restrict__ colpmax,
                   float* __restrict__ out) {
  __shared__ float satt[LQ];
  int b = blockIdx.x >> 2;
  int part = blockIdx.x & 3;
  int tid = threadIdx.x;
  if (part < 2) {
    for (int i = tid; i < LQ; i += 256) {
      size_t ri = ((size_t)b * LQ + i) * 2;
      float s = rowpsum[ri] + rowpsum[ri + 1];
      float m = fmaxf(rowpmax[ri], rowpmax[ri + 1]);
      satt[i] = __expf(m) / s;                      // max of softmax row
    }
    __syncthreads();
    int h = part * 256 + tid;
    const float* qb = q + (size_t)b * LQ * HD + h;
    float acc = 0.f;
    for (int i = 0; i < LQ; ++i) acc += satt[i] * qb[(size_t)i * HD];
    out[(size_t)b * HD + h] = acc;
  } else {
    for (int j = tid; j < LA; j += 256) {
      float s = 0.f, m = 0.f;
      for (int w = 0; w < 64; ++w) {                // fixed-order fold: deterministic
        size_t ci = ((size_t)b * 64 + w) * LA + j;
        s += colpsum[ci];
        m = fmaxf(m, colpmax[ci]);
      }
      satt[j] = __expf(m) / s;
    }
    __syncthreads();
    int h = (part - 2) * 256 + tid;
    const float* ab = a + (size_t)b * LA * HD + h;
    float acc = 0.f;
    for (int j = 0; j < LA; ++j) acc += satt[j] * ab[(size_t)j * HD];
    out[(size_t)NB * HD + (size_t)b * HD + h] = acc;
  }
}

extern "C" void kernel_launch(void* const* d_in, const int* in_sizes, int n_in,
                              void* d_out, int out_size, void* d_ws, size_t ws_size,
                              hipStream_t stream) {
  const float* q  = (const float*)d_in[0];
  const float* a  = (const float*)d_in[1];
  const float* U  = (const float*)d_in[2];
  const float* qm = (const float*)d_in[3];
  const float* am = (const float*)d_in[4];
  float* out = (float*)d_out;

  char* ws = (char*)d_ws;
  size_t off = 0;
  __bf16* a_bf   = (__bf16*)(ws + off); off += (size_t)NB * LA * HD * 2;   // 32 MB
  __bf16* qU_bf  = (__bf16*)(ws + off); off += (size_t)NB * LQ * HD * 2;   // 32 MB
  __bf16* Ut_bf  = (__bf16*)(ws + off); off += (size_t)HD * HD * 2;        // 0.5 MB
  float* rowpsum = (float*)(ws + off);  off += (size_t)NB * LQ * 2 * 4;    // 256 KB
  float* rowpmax = (float*)(ws + off);  off += (size_t)NB * LQ * 2 * 4;
  float* colpsum = (float*)(ws + off);  off += (size_t)NB * 64 * LA * 4;   // 8 MB
  float* colpmax = (float*)(ws + off);  off += (size_t)NB * 64 * LA * 4;

  hipLaunchKernelGGL(prep_kernel, dim3(2048), dim3(256), 0, stream, a, U, a_bf, Ut_bf);
  hipLaunchKernelGGL(gemm_qU_kernel, dim3(NB * (LQ / 32)), dim3(128), 0, stream,
                     q, Ut_bf, qU_bf);
  hipLaunchKernelGGL(score_kernel, dim3(NB * (LQ / 32)), dim3(128), 0, stream,
                     qU_bf, a_bf, qm, am, rowpsum, rowpmax, colpsum, colpmax);
  hipLaunchKernelGGL(finish_kernel, dim3(NB * 4), dim3(256), 0, stream,
                     q, a, rowpsum, rowpmax, colpsum, colpmax, out);
}
